// MultiHeadAttention_49864570307068
// MI455X (gfx1250) — matrix operand order repaired and checked by reading, not compiled, not run on hardware
//
#include <hip/hip_runtime.h>

// Problem constants (match reference).
constexpr int H    = 16;
constexpr int D    = 1024;
constexpr int LV   = 2048;
constexpr int HD   = 64;     // head dim
constexpr int BN   = 2;      // batch
constexpr int LQ   = 2048;
constexpr float LN_EPS = 1e-5f;

// Tiling
constexpr int MT   = 16;     // Q rows per block
constexpr int LVT  = 128;    // LV chunk per iteration (16 per wave x 8 waves)
constexpr int QS   = D + 8;  // LDS stride (bf16 elems); 2064B rows -> 16B aligned
constexpr int PS   = LVT + 8;// LDS stride for P tile; 272B rows -> 16B aligned

typedef __bf16 bf16_t;
typedef bf16_t v16bf __attribute__((ext_vector_type(16)));
typedef bf16_t v2bf  __attribute__((ext_vector_type(2)));
typedef float  v8f   __attribute__((ext_vector_type(8)));

union FragU {
    unsigned int u[8];
    uint4        q[2];
    v16bf        v;
};

// Hardware packed f32->bf16 conversion (v_cvt_pk_bf16_f32).
__device__ inline unsigned int pack_bf16x2(float lo, float hi) {
    v2bf p;
    p.x = (bf16_t)lo;
    p.y = (bf16_t)hi;
    return __builtin_bit_cast(unsigned int, p);
}

__device__ inline unsigned short cvt_bf16(float f) {
    return __builtin_bit_cast(unsigned short, (bf16_t)f);
}

// A-fragment (16x32 bf16, M x K) from LDS.
// Layout: lane half 0 -> K {0..15}, lane half 1 -> K {16..31} (matches B);
// each K-run of 8 bf16 is contiguous in the row -> two 16B LDS loads.
__device__ inline v16bf load_a_lds(const unsigned short* p, int stride) {
    const int lane  = threadIdx.x & 31;
    const int m     = lane & 15;
    const int kBase = (lane >> 4) * 16;
    FragU f;
    f.q[0] = *(const uint4*)(p + m * stride + kBase);        // K = kBase .. kBase+7
    f.q[1] = *(const uint4*)(p + m * stride + kBase + 8);    // K = kBase+8 .. kBase+15
    return f.v;
}

// B-fragment (32x16 bf16, K x N) straight from global fp32 with on-the-fly cvt.
// g points at element (k=0, n=0); row stride ldg (fp32 elems).
// Layout (contiguous k, matches A): lanes 0-15 N=lane K=0..15 (2/VGPR); lanes 16-31 N=lane-16 K=16..31.
__device__ inline v16bf load_b_g(const float* __restrict__ g, int ldg) {
    const int lane  = threadIdx.x & 31;
    const int n     = lane & 15;
    const int kBase = (lane >> 4) * 16;
    FragU f;
#pragma unroll
    for (int r = 0; r < 8; ++r) {
        long k = kBase + 2 * r;
        float lo = g[k * (long)ldg + n];
        float hi = g[(k + 1) * (long)ldg + n];
        f.u[r] = pack_bf16x2(lo, hi);
    }
    return f.v;
}

__device__ inline v8f wmma_bf16(v16bf a, v16bf b, v8f c) {
    return __builtin_amdgcn_wmma_f32_16x16x32_bf16(
        /*neg_a=*/false, a, /*neg_b=*/false, b,
        /*c_mod=*/(short)0, c, /*reuse_a=*/false, /*reuse_b=*/false);
}

__global__ __launch_bounds__(256)
void ext_attn_fused(const float* __restrict__ Vg,   // [B, LV, D]
                    const float* __restrict__ Qg,   // [B, LQ, D]
                    const float* __restrict__ Wa,   // [H, D, LV]
                    const float* __restrict__ ba,   // [H, LV]
                    const float* __restrict__ gamma,// [H, D]
                    const float* __restrict__ beta, // [H, D]
                    const float* __restrict__ Wp,   // [H, D, HD]
                    const float* __restrict__ bp,   // [H, HD]
                    float* __restrict__ out)        // [B, LQ, D]
{
    __shared__ unsigned short ldsQ[MT * QS];   // bf16 Q tile, later reused for normed
    __shared__ unsigned short ldsP[MT * PS];   // bf16 relu(QWa+ba) chunk
    __shared__ float red[2][MT];               // LN sum / sumsq

    const int tid  = threadIdx.x;
    const int lane = tid & 31;
    const int w    = tid >> 5;          // wave id 0..7
    const int n    = lane & 15;
    const int hiL  = lane >> 4;

    const int q0 = blockIdx.x * MT;
    const int h  = blockIdx.y;
    const int bi = blockIdx.z;

    const float* qTile = Qg + ((long)bi * LQ + q0) * D;
    const float* Wa_h  = Wa + (long)h * D * LV;
    const float* ba_h  = ba + (long)h * LV;
    const float* V_b   = Vg + (long)bi * LV * D;
    const float* g_h   = gamma + (long)h * D;
    const float* be_h  = beta  + (long)h * D;
    const float* Wp_h  = Wp + (long)h * D * HD;
    const float* bp_h  = bp + (long)h * HD;

    // --- Stage Q tile (16 x 1024) into LDS as bf16, pairs at a time ---
    for (int i = tid; i < MT * D / 2; i += 256) {
        int m = i >> 9;               // / (D/2)
        int d = (i & (D / 2 - 1)) * 2;
        const float* src = qTile + (long)m * D + d;
        *(unsigned int*)(ldsQ + m * QS + d) = pack_bf16x2(src[0], src[1]);
    }
    __syncthreads();

    // agg accumulators: this wave owns D columns [w*128, w*128+128) -> 8 C tiles
    v8f acc[8];
#pragma unroll
    for (int t = 0; t < 8; ++t) acc[t] = v8f{0.f, 0.f, 0.f, 0.f, 0.f, 0.f, 0.f, 0.f};

    for (int lv0 = 0; lv0 < LV; lv0 += LVT) {
        // --- Phase A: S(16x16) = Qtile(16x1024) @ Wa_h[:, lv0 + 16w] ---
        v8f s = v8f{0.f, 0.f, 0.f, 0.f, 0.f, 0.f, 0.f, 0.f};
        const float* waCol = Wa_h + lv0 + w * 16;
        for (int k0 = 0; k0 < D; k0 += 32) {
            v16bf af = load_a_lds(&ldsQ[k0], QS);
            v16bf bf = load_b_g(waCol + (long)k0 * LV, LV);
            s = wmma_bf16(af, bf, s);
        }
        // --- Phase B: bias + relu (v_max_num_f32), bf16 -> LDS P ---
        float bias = ba_h[lv0 + w * 16 + n];
#pragma unroll
        for (int r = 0; r < 8; ++r) {
            float x = fmaxf(s[r] + bias, 0.f);
            ldsP[(r + hiL * 8) * PS + w * 16 + n] = cvt_bf16(x);
        }
        __syncthreads();

        // --- Phase C: agg(16 x 128-cols-of-this-wave) += P(16x128) @ V[lv chunk] ---
#pragma unroll
        for (int kk = 0; kk < LVT; kk += 32) {
            v16bf af = load_a_lds(&ldsP[kk], PS);
            const float* vBase = V_b + (long)(lv0 + kk) * D + w * 128;
#pragma unroll
            for (int t = 0; t < 8; ++t) {
                v16bf bf = load_b_g(vBase + t * 16, D);
                acc[t] = wmma_bf16(af, bf, acc[t]);
            }
        }
        __syncthreads();   // protect ldsP before next iteration overwrites it
    }

    // --- Phase D: LayerNorm over D=1024 per row ---
    if (tid < 32) ((float*)red)[tid] = 0.f;
    __syncthreads();
#pragma unroll
    for (int r = 0; r < 8; ++r) {
        int m = r + hiL * 8;
        float sm = 0.f, sq = 0.f;
#pragma unroll
        for (int t = 0; t < 8; ++t) { float x = acc[t][r]; sm += x; sq += x * x; }
        atomicAdd(&red[0][m], sm);
        atomicAdd(&red[1][m], sq);
    }
    __syncthreads();
#pragma unroll
    for (int r = 0; r < 8; ++r) {
        int m = r + hiL * 8;
        float mu   = red[0][m] * (1.f / D);
        float var  = red[1][m] * (1.f / D) - mu * mu;
        float rstd = rsqrtf(var + LN_EPS);
#pragma unroll
        for (int t = 0; t < 8; ++t) {
            int d = w * 128 + t * 16 + n;
            float x = (acc[t][r] - mu) * rstd * g_h[d] + be_h[d];
            ldsQ[m * QS + d] = cvt_bf16(x);     // reuse Q buffer for normed (bf16)
        }
    }
    __syncthreads();

    // --- Phase E: heads(16x64) = normed(16x1024) @ Wp_h(1024x64) + bp ---
    if (w < 4) {
        v8f o = v8f{0.f, 0.f, 0.f, 0.f, 0.f, 0.f, 0.f, 0.f};
        const float* wpCol = Wp_h + w * 16;
        for (int k0 = 0; k0 < D; k0 += 32) {
            v16bf af = load_a_lds(&ldsQ[k0], QS);
            v16bf bf = load_b_g(wpCol + (long)k0 * HD, HD);
            o = wmma_bf16(af, bf, o);
        }
        float bias = bp_h[w * 16 + n];
#pragma unroll
        for (int r = 0; r < 8; ++r) {
            int m = r + hiL * 8;
            out[((long)bi * LQ + q0 + m) * D + h * HD + w * 16 + n] = o[r] + bias;
        }
    }
}

extern "C" void kernel_launch(void* const* d_in, const int* in_sizes, int n_in,
                              void* d_out, int out_size, void* d_ws, size_t ws_size,
                              hipStream_t stream) {
    const float* v     = (const float*)d_in[0];
    const float* q     = (const float*)d_in[1];
    const float* Wa    = (const float*)d_in[2];
    const float* ba    = (const float*)d_in[3];
    const float* gamma = (const float*)d_in[4];
    const float* beta  = (const float*)d_in[5];
    const float* Wp    = (const float*)d_in[6];
    const float* bp    = (const float*)d_in[7];
    float* out = (float*)d_out;

    dim3 grid(LQ / MT, H, BN);   // 128 x 16 x 2 blocks
    ext_attn_fused<<<grid, 256, 0, stream>>>(v, q, Wa, ba, gamma, beta, Wp, bp, out);
}
